// MossAttention_25735444038184
// MI455X (gfx1250) — matrix operand order repaired and checked by reading, not compiled, not run on hardware
//
#include <hip/hip_runtime.h>
#include <hip/hip_bf16.h>
#include <math.h>
#include <stdint.h>

#define B_   2
#define S_   2048
#define H_   4096
#define NH_  16
#define HD_  256
#define RD_  64
#define MTOT (B_ * S_)      // 4096 rows total
#define F3H  (3 * H_)       // 12288

typedef __attribute__((ext_vector_type(16))) __bf16 v16bf;
typedef __attribute__((ext_vector_type(8)))  float  v8f;
typedef __attribute__((ext_vector_type(4)))  unsigned int u32x4;
typedef __attribute__((ext_vector_type(8)))  int i32x8;
typedef __attribute__((ext_vector_type(4)))  int i32x4;

#if defined(__gfx1250__) && __has_builtin(__builtin_amdgcn_tensor_load_to_lds)
#define HAVE_TDM 1
#else
#define HAVE_TDM 0
#endif

// ---------------------------------------------------------------------------
// Fragment loaders matching CDNA5 WMMA VGPR layouts (wave32).
// A 16x32 bf16: lane m=lane&15; lanes 0-15 hold K {0..7,16..23}, lanes 16-31
// hold K {8..15,24..31} (ISA 7.12.2 16-bit A-matrix table).
// B 32x16 bf16: lane n=lane&15; lanes 0-15 hold K {0..7,16..23}, lanes 16-31
// hold K {8..15,24..31} (same K order as A).
// ---------------------------------------------------------------------------
__device__ inline v16bf load_a_frag(const __bf16* base, int ld) {
  const int lane = threadIdx.x & 31;
  const int m    = lane & 15;
  const int kh   = (lane >> 4) << 3;   // 0 or 8
  const __bf16* p = base + m * ld;
  v16bf a;
#pragma unroll
  for (int j = 0; j < 8; ++j) a[j]     = p[kh + j];
#pragma unroll
  for (int j = 0; j < 8; ++j) a[8 + j] = p[16 + kh + j];
  return a;
}

__device__ inline v16bf load_b_frag(const __bf16* base, int ld) {
  const int lane = threadIdx.x & 31;
  const int n    = lane & 15;
  const int kh   = (lane >> 4) << 3;   // 0 or 8
  const __bf16* p = base + n * ld;
  v16bf b;
#pragma unroll
  for (int j = 0; j < 8; ++j) b[j]     = p[kh + j];
#pragma unroll
  for (int j = 0; j < 8; ++j) b[8 + j] = p[16 + kh + j];
  return b;
}

// ---------------------------------------------------------------------------
// fp32 -> bf16 convert (grid-stride)
// ---------------------------------------------------------------------------
__global__ void cvt_f32_bf16(const float* __restrict__ in, __bf16* __restrict__ out, size_t n) {
  size_t i = (size_t)blockIdx.x * blockDim.x + threadIdx.x;
  const size_t stride = (size_t)gridDim.x * blockDim.x;
  for (; i < n; i += stride) out[i] = (__bf16)in[i];
}

// ---------------------------------------------------------------------------
// Tiled bf16 GEMM, fp32 accumulate. Block tile 64x128, K-chunk 32.
// 8 waves arranged 2(M) x 4(N), each owning a 32x32 tile (2x2 WMMA frags).
// Next-chunk global prefetch (global_prefetch_b8) issued before the barrier.
// ---------------------------------------------------------------------------
#define GM 64
#define GN 128
#define GK 32

template <bool OUT_BF16>
__global__ __launch_bounds__(256) void gemm_bf16_kernel(
    const __bf16* __restrict__ A, const __bf16* __restrict__ Bm,
    void* __restrict__ Cp, int M, int N, int K) {
  __shared__ __bf16 sA[GM][GK];    // 4 KB
  __shared__ __bf16 sBT[GN][GK];   // 8 KB, B staged transposed: [n][k]

  const int tid  = threadIdx.x;
  const int wave = tid >> 5;
  const int lane = tid & 31;
  const int bm = blockIdx.y * GM;
  const int bn = blockIdx.x * GN;
  const int wm = (wave >> 2) * 32;     // 0 / 32
  const int wn = (wave & 3) * 32;      // 0 / 32 / 64 / 96
  const int cn = lane & 15;
  const int cm = (lane >> 4) * 8;

  v8f acc[2][2] = {};

  for (int k0 = 0; k0 < K; k0 += GK) {
    // Stage A tile (64x32) row-major
    for (int i = tid; i < GM * GK; i += 256) {
      int r = i >> 5, c = i & 31;
      sA[r][c] = A[(size_t)(bm + r) * K + k0 + c];
    }
    // Stage B tile (32x128) transposed -> sBT[n][k]
    for (int i = tid; i < GK * GN; i += 256) {
      int r = i >> 7, c = i & 127;
      sBT[c][r] = Bm[(size_t)(k0 + r) * N + bn + c];
    }
    // Prefetch next K-chunk into cache while we compute on this one.
    if (k0 + GK < K) {
      __builtin_prefetch(&A[(size_t)(bm + (tid >> 2)) * K + k0 + GK + ((tid & 3) << 3)], 0, 3);
      __builtin_prefetch(&Bm[(size_t)(k0 + GK + (tid >> 3)) * N + bn + ((tid & 7) << 4)], 0, 3);
    }
    __syncthreads();

#pragma unroll
    for (int mi = 0; mi < 2; ++mi) {
      v16bf af = load_a_frag(&sA[wm + mi * 16][0], GK);
#pragma unroll
      for (int ni = 0; ni < 2; ++ni) {
        v16bf bf = load_b_frag(&sBT[wn + ni * 16][0], GK);
        acc[mi][ni] = __builtin_amdgcn_wmma_f32_16x16x32_bf16(
            false, af, false, bf, (short)0, acc[mi][ni], false, false);
      }
    }
    __syncthreads();
  }

  // Epilogue: C layout — lane n = lane&15, VGPR i -> row cm+i
#pragma unroll
  for (int mi = 0; mi < 2; ++mi) {
#pragma unroll
    for (int ni = 0; ni < 2; ++ni) {
#pragma unroll
      for (int i = 0; i < 8; ++i) {
        const size_t r = (size_t)(bm + wm + mi * 16 + cm + i);
        const size_t c = (size_t)(bn + wn + ni * 16 + cn);
        if constexpr (OUT_BF16)
          ((__bf16*)Cp)[r * N + c] = (__bf16)acc[mi][ni][i];
        else
          ((float*)Cp)[r * N + c] = acc[mi][ni][i];
      }
    }
  }
}

// ---------------------------------------------------------------------------
// GPT-J interleaved RoPE on Q (col h*HD+2i) and K (col 2H + h*HD+2i), in place
// on the bf16 QKV buffer. One thread per (row, head, pair).
// ---------------------------------------------------------------------------
__global__ void rope_kernel(__bf16* __restrict__ qkv, const int* __restrict__ pos) {
  const int total = MTOT * NH_ * (RD_ / 2);
  int idx = blockIdx.x * 256 + threadIdx.x;
  if (idx >= total) return;
  const int i    = idx & 31;          // pair index 0..31
  const int h    = (idx >> 5) & 15;
  const int srow = idx >> 9;          // flat (b*S + s)

  const float p   = (float)pos[srow];
  const float inv = __powf(10000.0f, -(2.0f * (float)i) / (float)RD_);
  const float fr  = p * inv;
  const float c = __cosf(fr), s = __sinf(fr);

  const size_t base = (size_t)srow * F3H;
  int col = h * HD_ + 2 * i;                 // Q
  {
    float x1 = (float)qkv[base + col], x2 = (float)qkv[base + col + 1];
    qkv[base + col]     = (__bf16)(x1 * c - x2 * s);
    qkv[base + col + 1] = (__bf16)(x2 * c + x1 * s);
  }
  col += 2 * H_;                             // K (reference split order q,v,k)
  {
    float x1 = (float)qkv[base + col], x2 = (float)qkv[base + col + 1];
    qkv[base + col]     = (__bf16)(x1 * c - x2 * s);
    qkv[base + col + 1] = (__bf16)(x2 * c + x1 * s);
  }
}

// ---------------------------------------------------------------------------
// Flash attention: block = (64 q-rows, head, batch), 256 threads (8 waves).
// Key tiles of 32, online softmax in LDS. Scores: 4x2 tiles of 16x16, one per
// wave, contraction 256 via 8 WMMAs. PV: each wave owns 16 rows x 128 cols
// (8 WMMA tiles), V staged transposed so B-frags are contiguous.
// K tile staged by the Tensor Data Mover (tensor_load_to_lds + TENSORcnt)
// when the builtin is available; V staged manually (needs transpose).
// ---------------------------------------------------------------------------
#define QT 64
#define KT 32

__global__ __launch_bounds__(256) void flash_attn_kernel(
    const __bf16* __restrict__ qkv, __bf16* __restrict__ attn) {
  __shared__ __bf16 sQ[QT][HD_];    // 32 KB
  __shared__ __bf16 sK[KT][HD_];    // 16 KB
  __shared__ __bf16 sVT[HD_][KT];   // 16 KB (V transposed: [hd][key])
  __shared__ float  sS[QT][KT];     // 8 KB
  __shared__ __bf16 sP[QT][KT];     // 4 KB
  __shared__ float  sM[QT], sL[QT], sAlpha[QT];

  const int tid  = threadIdx.x;
  const int wave = tid >> 5;
  const int lane = tid & 31;
  const int q0 = blockIdx.x * QT;
  const int h  = blockIdx.y;
  const int b  = blockIdx.z;
  const size_t rowbase = (size_t)b * S_;
  const int qcol = h * HD_;
  const int vcol = H_ + h * HD_;
  const int kcol = 2 * H_ + h * HD_;

  for (int i = tid; i < QT * HD_; i += 256) {
    int r = i >> 8, c = i & (HD_ - 1);
    sQ[r][c] = qkv[(rowbase + q0 + r) * (size_t)F3H + qcol + c];
  }
  if (tid < QT) { sM[tid] = -1e30f; sL[tid] = 0.0f; }

  const int sm = (wave >> 1) * 16;       // score/O m-tile
  const int sn = (wave & 1) * 16;        // score n-tile
  const int on = (wave & 1) * 128;       // O n-half
  const int cn = lane & 15;
  const int cm = (lane >> 4) * 8;

  v8f oacc[8] = {};
  const float scale = 0.0625f;           // 256^-0.5

  const int kmax = q0 + QT;
  for (int k0 = 0; k0 < kmax; k0 += KT) {
    __syncthreads();   // protect sK/sVT from previous iteration's readers

#if HAVE_TDM
    // --- TDM: DMA the 32x256 bf16 K tile (row stride F3H elems) into sK ---
    if (wave == 0) {
      const uint64_t ga = (uint64_t)(uintptr_t)(qkv + (rowbase + k0) * (size_t)F3H + kcol);
      const uint32_t lds = (uint32_t)(uintptr_t)&sK[0][0];   // LDS offset = addr[31:0]
      u32x4 g0;
      g0[0] = 1u;                                            // count=1, no gather
      g0[1] = lds;                                           // lds_addr
      g0[2] = (uint32_t)ga;                                  // global_addr[31:0]
      g0[3] = (uint32_t)((ga >> 32) & 0x1FFFFFFu) | (2u << 30); // addr[56:32] | type=2
      i32x8 g1;
      g1[0] = (1 << 16);                                     // data_size=1 (2 bytes)
      g1[1] = (int)((F3H & 0xFFFF) << 16);                   // tensor_dim0[15:0]
      g1[2] = (int)((F3H >> 16) | ((MTOT & 0xFFFF) << 16));  // dim0[31:16] | dim1[15:0]
      g1[3] = (int)((MTOT >> 16) | (HD_ << 16));             // dim1[31:16] | tile_dim0
      g1[4] = KT;                                            // tile_dim1 | tile_dim2=0
      g1[5] = F3H;                                           // tensor_dim0_stride[31:0]
      g1[6] = 0;                                             // stride hi | dim1_stride lo
      g1[7] = 0;
      i32x4 gz = {0, 0, 0, 0};
#if __clang_major__ >= 23
      i32x8 gz8 = {0, 0, 0, 0, 0, 0, 0, 0};
      __builtin_amdgcn_tensor_load_to_lds(g0, g1, gz, gz, gz8, 0);
#else
      __builtin_amdgcn_tensor_load_to_lds(g0, g1, gz, gz, 0);
#endif
      __builtin_amdgcn_s_wait_tensorcnt(0);
    }
    // V tile: manual staging (must land transposed in LDS)
    for (int i = tid; i < KT * HD_; i += 256) {
      int r = i >> 8, c = i & (HD_ - 1);
      sVT[c][r] = qkv[(rowbase + k0 + r) * (size_t)F3H + vcol + c];
    }
#else
    for (int i = tid; i < KT * HD_; i += 256) {
      int r = i >> 8, c = i & (HD_ - 1);
      const size_t row = (rowbase + k0 + r) * (size_t)F3H;
      sK[r][c]  = qkv[row + kcol + c];
      sVT[c][r] = qkv[row + vcol + c];
    }
#endif
    __syncthreads();

    // scores S = Q K^T : both frags read row-major rows (K^T column = K row)
    v8f sacc = {};
#pragma unroll
    for (int c = 0; c < HD_; c += 32) {
      v16bf aq = load_a_frag(&sQ[sm][c], HD_);
      v16bf bk = load_b_frag(&sK[sn][c], HD_);
      sacc = __builtin_amdgcn_wmma_f32_16x16x32_bf16(
          false, aq, false, bk, (short)0, sacc, false, false);
    }
#pragma unroll
    for (int i = 0; i < 8; ++i)
      sS[sm + cm + i][sn + cn] = sacc[i] * scale;
    __syncthreads();

    // online softmax: one thread per q-row
    if (tid < QT) {
      const int qi = q0 + tid;
      float mprev = sM[tid], mnew = mprev;
#pragma unroll 8
      for (int j = 0; j < KT; ++j) {
        float sv = ((k0 + j) <= qi) ? sS[tid][j] : -1e30f;
        sS[tid][j] = sv;
        mnew = fmaxf(mnew, sv);
      }
      const float alpha = __expf(mprev - mnew);
      float lsum = 0.0f;
#pragma unroll 8
      for (int j = 0; j < KT; ++j) {
        float pj = __expf(sS[tid][j] - mnew);
        sP[tid][j] = (__bf16)pj;
        lsum += pj;
      }
      sM[tid] = mnew;
      sL[tid] = sL[tid] * alpha + lsum;
      sAlpha[tid] = alpha;
    }
    __syncthreads();

    // rescale running O and accumulate P V
    float al[8];
#pragma unroll
    for (int i = 0; i < 8; ++i) al[i] = sAlpha[sm + cm + i];
    v16bf ap = load_a_frag(&sP[sm][0], KT);
#pragma unroll
    for (int ni = 0; ni < 8; ++ni) {
#pragma unroll
      for (int i = 0; i < 8; ++i) oacc[ni][i] *= al[i];
      v16bf bv = load_b_frag(&sVT[on + ni * 16][0], KT);
      oacc[ni] = __builtin_amdgcn_wmma_f32_16x16x32_bf16(
          false, ap, false, bv, (short)0, oacc[ni], false, false);
    }
  }
  __syncthreads();

  float linv[8];
#pragma unroll
  for (int i = 0; i < 8; ++i) linv[i] = 1.0f / sL[sm + cm + i];
#pragma unroll
  for (int ni = 0; ni < 8; ++ni) {
#pragma unroll
    for (int i = 0; i < 8; ++i) {
      const size_t row = rowbase + q0 + sm + cm + i;
      attn[row * (size_t)H_ + qcol + on + ni * 16 + cn] =
          (__bf16)(oacc[ni][i] * linv[i]);
    }
  }
}

// ---------------------------------------------------------------------------
// Launch: converts -> QKV GEMM (bf16 out) -> RoPE -> flash attn -> out-proj.
// Workspace carve (bf16): X 32MiB | Wqkv 96MiB | Wout 32MiB | QKV 96MiB |
// ATTN 32MiB  (288 MiB total).
// ---------------------------------------------------------------------------
extern "C" void kernel_launch(void* const* d_in, const int* in_sizes, int n_in,
                              void* d_out, int out_size, void* d_ws, size_t ws_size,
                              hipStream_t stream) {
  const float* hidden = (const float*)d_in[0];
  const int*   pos    = (const int*)d_in[1];
  const float* w_qkv  = (const float*)d_in[2];
  const float* w_out  = (const float*)d_in[3];
  float* out = (float*)d_out;

  __bf16* Xb    = (__bf16*)d_ws;
  __bf16* Wqkvb = Xb    + (size_t)MTOT * H_;
  __bf16* Woutb = Wqkvb + (size_t)H_ * F3H;
  __bf16* QKVb  = Woutb + (size_t)H_ * H_;
  __bf16* ATTNb = QKVb  + (size_t)MTOT * F3H;

  cvt_f32_bf16<<<1024, 256, 0, stream>>>(hidden, Xb,    (size_t)MTOT * H_);
  cvt_f32_bf16<<<2048, 256, 0, stream>>>(w_qkv,  Wqkvb, (size_t)H_ * F3H);
  cvt_f32_bf16<<<1024, 256, 0, stream>>>(w_out,  Woutb, (size_t)H_ * H_);

  dim3 g1(F3H / GN, MTOT / GM);
  gemm_bf16_kernel<true><<<g1, 256, 0, stream>>>(Xb, Wqkvb, (void*)QKVb, MTOT, F3H, H_);

  const int pairs = MTOT * NH_ * (RD_ / 2);
  rope_kernel<<<(pairs + 255) / 256, 256, 0, stream>>>(QKVb, pos);

  dim3 ga(S_ / QT, NH_, B_);
  flash_attn_kernel<<<ga, 256, 0, stream>>>(QKVb, ATTNb);

  dim3 g2(H_ / GN, MTOT / GM);
  gemm_bf16_kernel<false><<<g2, 256, 0, stream>>>(ATTNb, Woutb, (void*)out, MTOT, H_, H_);
}
